// GPT_43112881717694
// MI455X (gfx1250) — matrix operand order repaired and checked by reading, not compiled, not run on hardware
//
#include <hip/hip_runtime.h>
#include <hip/hip_bf16.h>

typedef __attribute__((ext_vector_type(8)))  __bf16 v8bf;
typedef __attribute__((ext_vector_type(16))) __bf16 v16bf;
typedef __attribute__((ext_vector_type(8)))  float  v8f;

#define CDIM 256
#define TT   1024
#define NH   8
#define HD   32
#define NROWS 32768   // B*T

// ---------------------------------------------------------------------------
// TDM (Tensor Data Mover) support, gated so compile can never break.
// ---------------------------------------------------------------------------
#if defined(__AMDGCN__)
#if __has_builtin(__builtin_amdgcn_tensor_load_to_lds)
#define USE_TDM 1
typedef __attribute__((ext_vector_type(4))) unsigned int u32x4;
typedef __attribute__((ext_vector_type(8))) int i32x8;
typedef __attribute__((ext_vector_type(4))) int i32x4;

// 2D tile load: tile (tl0 x tl1) of a (td0 x td1) bf16 tensor with row stride
// 'stride' (elements), from global 'gp' (tile origin) into LDS offset lds_off.
// D# packing per CDNA5 ISA ch.8.
static __device__ __forceinline__ void tdm_load_2d(unsigned lds_off, const void* gp,
                                                   unsigned td0, unsigned td1,
                                                   unsigned tl0, unsigned tl1,
                                                   unsigned long long stride) {
  unsigned long long ga = (unsigned long long)(size_t)gp;
  u32x4 g0 = {1u, lds_off, (unsigned)ga,
              ((unsigned)((ga >> 32) & 0x01ffffffull)) | 0x80000000u};  // type=2
  i32x8 g1;
  g1[0] = 0x00010000;                               // data_size = 2 bytes
  g1[1] = (int)((td0 & 0xffffu) << 16);             // tensor_dim0[15:0]
  g1[2] = (int)((td0 >> 16) | ((td1 & 0xffffu) << 16));
  g1[3] = (int)((td1 >> 16) | (tl0 << 16));         // tile_dim0
  g1[4] = (int)(tl1 & 0xffffu);                     // tile_dim1 (tile_dim2 = 0)
  g1[5] = (int)(stride & 0xffffffffull);            // tensor_dim0_stride lo
  g1[6] = (int)((stride >> 32) & 0xffffull);
  g1[7] = 0;
  i32x4 z4 = {0, 0, 0, 0};
#if __has_include(<hip/amd_detail/amd_gfx1250_TDM.h>)
  i32x8 z8 = {0, 0, 0, 0, 0, 0, 0, 0};
  __builtin_amdgcn_tensor_load_to_lds(g0, g1, z4, z4, z8, 0);
#else
  __builtin_amdgcn_tensor_load_to_lds(g0, g1, z4, z4, 0);
#endif
}
#endif

// Async global->LDS copy (ASYNCcnt path), gated.  Builtin signature (from
// compiler diagnostic): (v4i32 AS1*, v4i32 AS3*, int offset, int cpol).
#if __has_builtin(__builtin_amdgcn_global_load_async_to_lds_b128)
#define USE_ASYNC 1
typedef int v4i_t __attribute__((vector_size(16)));
typedef __attribute__((address_space(1))) v4i_t* as1v4;
typedef __attribute__((address_space(3))) v4i_t* as3v4;
static __device__ __forceinline__ void async_cp16(const void* g, void* l) {
  __builtin_amdgcn_global_load_async_to_lds_b128(
      (as1v4)(unsigned long long)(size_t)g, (as3v4)(unsigned)(size_t)l, 0, 0);
}
#endif
#endif  // __AMDGCN__

// ---------------------------------------------------------------------------
// WMMA fragment helpers (layouts per CDNA5 ISA 7.12.2; LDS row stride = 32)
// ---------------------------------------------------------------------------
static __device__ __forceinline__ v16bf cat8(v8bf lo, v8bf hi) {
  return __builtin_shufflevector(lo, hi, 0,1,2,3,4,5,6,7,8,9,10,11,12,13,14,15);
}
static __device__ __forceinline__ v16bf ldsA_frag(const __bf16* rowp, int half) {
  v8bf lo = *(const v8bf*)(rowp + half * 8);
  v8bf hi = *(const v8bf*)(rowp + 16 + half * 8);
  return cat8(lo, hi);
}
static __device__ __forceinline__ v16bf ldsB_frag(const __bf16* rowp, int half) {
  v8bf lo = *(const v8bf*)(rowp + half * 8);
  v8bf hi = *(const v8bf*)(rowp + 16 + half * 8);
  return cat8(lo, hi);
}

// ---------------------------------------------------------------------------
__global__ __launch_bounds__(256) void embed_kernel(const int* __restrict__ idx,
                                                    const float* __restrict__ tok,
                                                    const float* __restrict__ pos,
                                                    float* __restrict__ x) {
  int row = blockIdx.x, c = threadIdx.x;
  int t = row & (TT - 1);
  int id = idx[row];
  x[(size_t)row * CDIM + c] = tok[(size_t)id * CDIM + c] + pos[(size_t)t * CDIM + c];
}

// Weight convert + transpose: W[K,N] f32 -> Wt[N,K] bf16  (once per call)
__global__ __launch_bounds__(256) void wcvt_kernel(const float* __restrict__ W,
                                                   __bf16* __restrict__ Wt,
                                                   int K, int N) {
  int idx = blockIdx.x * 256 + threadIdx.x;
  int k = idx / N, n = idx - k * N;
  Wt[(size_t)n * K + k] = (__bf16)W[idx];
}

// LayerNorm over C=256: fp32 in, bf16 out
__global__ __launch_bounds__(256) void ln_kernel(const float* __restrict__ x,
                                                 const float* __restrict__ g,
                                                 const float* __restrict__ bb,
                                                 __bf16* __restrict__ o) {
  __shared__ float red[256];
  int row = blockIdx.x, t = threadIdx.x;
  float v = x[(size_t)row * CDIM + t];
  red[t] = v;
  __syncthreads();
  for (int s = 128; s > 0; s >>= 1) {
    if (t < s) red[t] += red[t + s];
    __syncthreads();
  }
  float mean = red[0] * (1.0f / 256.0f);
  __syncthreads();
  float d = v - mean;
  red[t] = d * d;
  __syncthreads();
  for (int s = 128; s > 0; s >>= 1) {
    if (t < s) red[t] += red[t + s];
    __syncthreads();
  }
  float var = red[0] * (1.0f / 256.0f);
  float r = rsqrtf(var + 1e-5f);
  o[(size_t)row * CDIM + t] = (__bf16)(d * r * g[t] + bb[t]);
}

// ---------------------------------------------------------------------------
// GEMM: C[M,N] = A[M,K] * Wt[N,K]^T + bias.  A,Wt bf16; fp32 accum via WMMA.
// 256 threads = 8 waves; tile 128x128, K-step 32; wave -> 32x64 (2x4 WMMA).
// TDM double-buffered LDS staging (fallback: cooperative b128 copies).
// ---------------------------------------------------------------------------
template <int EPI, int OUTBF>
__global__ __launch_bounds__(256) void gemm_kernel(const __bf16* __restrict__ A,
                                                   const __bf16* __restrict__ Wt,
                                                   const float* __restrict__ bias,
                                                   const float* __restrict__ R,
                                                   void* __restrict__ Cout,
                                                   int M, int K, int N) {
  __shared__ __bf16 sA[2][128][32];
  __shared__ __bf16 sB[2][128][32];

  const int tid = threadIdx.x;
  const int lane = tid & 31, wave = tid >> 5;
  const int wm = wave >> 1, wn = wave & 1;
  const int half = lane >> 4, l16 = lane & 15;
  const int m0 = blockIdx.x * 128, n0 = blockIdx.y * 128;

  const v8f zv = {0.f, 0.f, 0.f, 0.f, 0.f, 0.f, 0.f, 0.f};
  v8f acc[2][4];
#pragma unroll
  for (int i = 0; i < 2; ++i)
#pragma unroll
    for (int j = 0; j < 4; ++j) acc[i][j] = zv;

  auto stage = [&](int buf, int k0) {
#if defined(USE_TDM)
    if (wave == 0) {
      tdm_load_2d((unsigned)(size_t)&sA[buf][0][0], &A[(size_t)m0 * K + k0],
                  (unsigned)K, (unsigned)M, 32u, 128u, (unsigned long long)K);
      tdm_load_2d((unsigned)(size_t)&sB[buf][0][0], &Wt[(size_t)n0 * K + k0],
                  (unsigned)K, (unsigned)N, 32u, 128u, (unsigned long long)K);
    }
#else
#pragma unroll
    for (int i = 0; i < 2; ++i) {
      int idx = tid + i * 256;
      int r = idx >> 2, c = (idx & 3) * 8;
      *(v8bf*)&sA[buf][r][c] = *(const v8bf*)&A[(size_t)(m0 + r) * K + k0 + c];
      *(v8bf*)&sB[buf][r][c] = *(const v8bf*)&Wt[(size_t)(n0 + r) * K + k0 + c];
    }
    if (k0 + 32 < K)
      __builtin_prefetch(&A[(size_t)(m0 + (tid >> 1)) * K + k0 + 32 + (tid & 1) * 16], 0, 1);
#endif
  };

  stage(0, 0);
  const int nk = K >> 5;
  for (int kk = 0; kk < nk; ++kk) {
    const int buf = kk & 1;
#if defined(USE_TDM)
    __builtin_amdgcn_s_wait_tensorcnt(0);
#endif
    __syncthreads();                       // staged tile visible to all waves
    if (kk + 1 < nk) stage(buf ^ 1, (kk + 1) * 32);

    v16bf af[2], bfr[4];
#pragma unroll
    for (int t = 0; t < 2; ++t) af[t] = ldsA_frag(&sA[buf][wm * 32 + t * 16 + l16][0], half);
#pragma unroll
    for (int t = 0; t < 4; ++t) bfr[t] = ldsA_frag(&sB[buf][wn * 64 + t * 16 + l16][0], half);
#pragma unroll
    for (int i = 0; i < 2; ++i)
#pragma unroll
      for (int j = 0; j < 4; ++j)
        acc[i][j] = __builtin_amdgcn_wmma_f32_16x16x32_bf16(
            false, af[i], false, bfr[j], (short)0, acc[i][j], false, false);
    __syncthreads();                       // all reads done before buf reuse
  }

#pragma unroll
  for (int i = 0; i < 2; ++i)
#pragma unroll
    for (int j = 0; j < 4; ++j)
#pragma unroll
      for (int v = 0; v < 8; ++v) {
        int row = m0 + wm * 32 + i * 16 + half * 8 + v;
        int col = n0 + wn * 64 + j * 16 + l16;
        float val = acc[i][j][v];
        if (bias) val += bias[col];
        if (EPI == 1) val += R[(size_t)row * N + col];
        if (EPI == 2) {
          float u = val;
          val = 0.5f * u * (1.0f + tanhf(0.7978845608028654f * (u + 0.044715f * u * u * u)));
        }
        if (OUTBF) ((__bf16*)Cout)[(size_t)row * N + col] = (__bf16)val;
        else       ((float*)Cout)[(size_t)row * N + col] = val;
      }
}

// ---------------------------------------------------------------------------
// Causal attention over bf16 qkv[M, 768].  Block = 4 waves, 64 q-rows per
// block (16 per wave).  Online softmax; WMMA for Q*K^T and P*V.
// Fast (mask-free) path for fully-causal kv blocks; masked path only on the
// <=2 diagonal blocks per wave.
// ---------------------------------------------------------------------------
__global__ __launch_bounds__(128) void attn_kernel(const __bf16* __restrict__ qkv,
                                                   __bf16* __restrict__ y) {
  __shared__ __bf16 sK[32][32];      // [kv][d]
  __shared__ __bf16 sVt[32][32];     // [d][kv]
  __shared__ __bf16 sP[4][16][32];   // per-wave P restage (C-layout -> A-layout)

  const int tid = threadIdx.x;
  const int lane = tid & 31, wave = tid >> 5;
  const int half = lane >> 4, l16 = lane & 15;
  const int bq = blockIdx.x, h = blockIdx.y, b = blockIdx.z;
  const int qbase = bq * 64 + wave * 16;
  const float scale = 0.17677669529663687f;  // 1/sqrt(32), folded post-WMMA

  const __bf16* qrow = &qkv[(size_t)(b * TT + qbase + l16) * 768 + h * HD];
  v16bf qf = cat8(*(const v8bf*)(qrow + half * 8), *(const v8bf*)(qrow + 16 + half * 8));

  const v8f zv = {0.f, 0.f, 0.f, 0.f, 0.f, 0.f, 0.f, 0.f};
  v8f o0 = zv, o1 = zv;
  float m8[8], l8[8];
#pragma unroll
  for (int v = 0; v < 8; ++v) { m8[v] = -3.0e38f; l8[v] = 0.0f; }

  const int kv_hi = bq * 64 + 64;
  for (int kv0 = 0; kv0 < kv_hi; kv0 += 32) {
    {  // cooperative stage: K row-major (async if available), V transposed
      int r = tid >> 2, c = (tid & 3) * 8;  // 128 threads cover 32x32 exactly
      const __bf16* kp = &qkv[(size_t)(b * TT + kv0 + r) * 768 + CDIM + h * HD + c];
#if defined(USE_ASYNC)
      async_cp16(kp, &sK[r][c]);
#else
      *(v8bf*)&sK[r][c] = *(const v8bf*)kp;
#endif
      v8bf vv = *(const v8bf*)(kp + CDIM);
#pragma unroll
      for (int j = 0; j < 8; ++j) sVt[c + j][r] = vv[j];
    }
#if defined(USE_ASYNC)
    asm volatile("s_wait_asynccnt 0" ::: "memory");
#endif
    __syncthreads();

    if (kv0 <= qbase + 15) {  // wave-uniform: this step has active columns
      v16bf kf0 = ldsB_frag(&sK[l16][0], half);
      v16bf kf1 = ldsB_frag(&sK[16 + l16][0], half);
      v8f s0 = __builtin_amdgcn_wmma_f32_16x16x32_bf16(false, qf, false, kf0, (short)0, zv, false, false);
      v8f s1 = __builtin_amdgcn_wmma_f32_16x16x32_bf16(false, qf, false, kf1, (short)0, zv, false, false);

      if (kv0 + 31 <= qbase) {
        // ---- fast path: fully unmasked block, no compares / guards ----
#pragma unroll
        for (int v = 0; v < 8; ++v) {
          float sv0 = s0[v] * scale;
          float sv1 = s1[v] * scale;
          float mx = fmaxf(sv0, sv1);
          mx = fmaxf(mx, __shfl_xor(mx, 1, 32));
          mx = fmaxf(mx, __shfl_xor(mx, 2, 32));
          mx = fmaxf(mx, __shfl_xor(mx, 4, 32));
          mx = fmaxf(mx, __shfl_xor(mx, 8, 32));
          float mnew = fmaxf(m8[v], mx);
          float sf = __expf(m8[v] - mnew);
          m8[v] = mnew;
          float p0 = __expf(sv0 - mnew);
          float p1 = __expf(sv1 - mnew);
          float rs = p0 + p1;
          rs += __shfl_xor(rs, 1, 32);
          rs += __shfl_xor(rs, 2, 32);
          rs += __shfl_xor(rs, 4, 32);
          rs += __shfl_xor(rs, 8, 32);
          l8[v] = l8[v] * sf + rs;
          o0[v] *= sf;
          o1[v] *= sf;
          sP[wave][half * 8 + v][l16] = (__bf16)p0;
          sP[wave][half * 8 + v][16 + l16] = (__bf16)p1;
        }
      } else {
        // ---- masked path: diagonal block(s) ----
#pragma unroll
        for (int v = 0; v < 8; ++v) {
          int rowg = qbase + half * 8 + v;
          float sv0 = (kv0 + l16 <= rowg) ? s0[v] * scale : -3.0e38f;
          float sv1 = (kv0 + 16 + l16 <= rowg) ? s1[v] * scale : -3.0e38f;
          float mx = fmaxf(sv0, sv1);
          mx = fmaxf(mx, __shfl_xor(mx, 1, 32));
          mx = fmaxf(mx, __shfl_xor(mx, 2, 32));
          mx = fmaxf(mx, __shfl_xor(mx, 4, 32));
          mx = fmaxf(mx, __shfl_xor(mx, 8, 32));
          float mnew = fmaxf(m8[v], mx);
          float sf = __expf(m8[v] - mnew);
          m8[v] = mnew;
          float p0 = (sv0 > -1.0e30f) ? __expf(sv0 - mnew) : 0.0f;
          float p1 = (sv1 > -1.0e30f) ? __expf(sv1 - mnew) : 0.0f;
          float rs = p0 + p1;
          rs += __shfl_xor(rs, 1, 32);
          rs += __shfl_xor(rs, 2, 32);
          rs += __shfl_xor(rs, 4, 32);
          rs += __shfl_xor(rs, 8, 32);
          l8[v] = l8[v] * sf + rs;
          o0[v] *= sf;
          o1[v] *= sf;
          sP[wave][half * 8 + v][l16] = (__bf16)p0;
          sP[wave][half * 8 + v][16 + l16] = (__bf16)p1;
        }
      }
      asm volatile("s_wait_dscnt 0" ::: "memory");  // wave-private cross-lane RAW
      v16bf pf = ldsA_frag(&sP[wave][l16][0], half);
      v16bf vf0 = ldsB_frag(&sVt[l16][0], half);
      v16bf vf1 = ldsB_frag(&sVt[16 + l16][0], half);
      o0 = __builtin_amdgcn_wmma_f32_16x16x32_bf16(false, pf, false, vf0, (short)0, o0, false, false);
      o1 = __builtin_amdgcn_wmma_f32_16x16x32_bf16(false, pf, false, vf1, (short)0, o1, false, false);
    }
    __syncthreads();
  }

#pragma unroll
  for (int v = 0; v < 8; ++v) {
    float inv = 1.0f / l8[v];
    size_t base = (size_t)(b * TT + qbase + half * 8 + v) * CDIM + h * HD;
    y[base + l16] = (__bf16)(o0[v] * inv);
    y[base + 16 + l16] = (__bf16)(o1[v] * inv);
  }
}

// ---------------------------------------------------------------------------
extern "C" void kernel_launch(void* const* d_in, const int* in_sizes, int n_in,
                              void* d_out, int out_size, void* d_ws, size_t ws_size,
                              hipStream_t stream) {
  (void)in_sizes; (void)n_in; (void)out_size; (void)ws_size;
  const int*   idx   = (const int*)d_in[0];
  const float* tok   = (const float*)d_in[1];
  const float* pos   = (const float*)d_in[2];
  const float* ln1g  = (const float*)d_in[3];
  const float* ln1b  = (const float*)d_in[4];
  const float* wqkv  = (const float*)d_in[5];
  const float* bqkv  = (const float*)d_in[6];
  const float* wo    = (const float*)d_in[7];
  const float* bo    = (const float*)d_in[8];
  const float* ln2g  = (const float*)d_in[9];
  const float* ln2b  = (const float*)d_in[10];
  const float* wfc   = (const float*)d_in[11];
  const float* bfc   = (const float*)d_in[12];
  const float* wpr   = (const float*)d_in[13];
  const float* bpr   = (const float*)d_in[14];
  const float* lnfg  = (const float*)d_in[15];
  const float* lnfb  = (const float*)d_in[16];
  const float* wlm   = (const float*)d_in[17];
  float* out = (float*)d_out;

  char* p = (char*)d_ws;
  float*  x    = (float*)p;   p += (size_t)NROWS * CDIM * 4;
  __bf16* hbuf = (__bf16*)p;  p += (size_t)NROWS * CDIM * 2;
  __bf16* qkvb = (__bf16*)p;  p += (size_t)NROWS * 3 * CDIM * 2;
  __bf16* ybuf = (__bf16*)p;  p += (size_t)NROWS * CDIM * 2;
  __bf16* ffb  = (__bf16*)p;  p += (size_t)NROWS * 1024 * 2;
  __bf16* qkvT = (__bf16*)p;  p += (size_t)8 * 768 * 256 * 2;
  __bf16* woT  = (__bf16*)p;  p += (size_t)8 * 256 * 256 * 2;
  __bf16* wfcT = (__bf16*)p;  p += (size_t)8 * 1024 * 256 * 2;
  __bf16* wprT = (__bf16*)p;  p += (size_t)8 * 256 * 1024 * 2;
  __bf16* wlmT = (__bf16*)p;

  // per-call weight convert+transpose into bf16 arena (~12.7MB, L2-resident)
  for (int l = 0; l < 8; ++l) {
    wcvt_kernel<<<(256 * 768) / 256, 256, 0, stream>>>(wqkv + (size_t)l * 256 * 768,
                                                       qkvT + (size_t)l * 768 * 256, 256, 768);
    wcvt_kernel<<<(256 * 256) / 256, 256, 0, stream>>>(wo + (size_t)l * 65536,
                                                       woT + (size_t)l * 65536, 256, 256);
    wcvt_kernel<<<(256 * 1024) / 256, 256, 0, stream>>>(wfc + (size_t)l * 262144,
                                                        wfcT + (size_t)l * 262144, 256, 1024);
    wcvt_kernel<<<(1024 * 256) / 256, 256, 0, stream>>>(wpr + (size_t)l * 262144,
                                                        wprT + (size_t)l * 262144, 1024, 256);
  }
  wcvt_kernel<<<(256 * 128) / 256, 256, 0, stream>>>(wlm, wlmT, 256, 128);

  embed_kernel<<<NROWS, 256, 0, stream>>>(idx, tok, pos, x);

  for (int l = 0; l < 8; ++l) {
    ln_kernel<<<NROWS, 256, 0, stream>>>(x, ln1g + l * CDIM, ln1b + l * CDIM, hbuf);
    gemm_kernel<0, 1><<<dim3(NROWS / 128, 6), 256, 0, stream>>>(
        hbuf, qkvT + (size_t)l * 768 * 256, bqkv + (size_t)l * 3 * CDIM,
        nullptr, qkvb, NROWS, CDIM, 3 * CDIM);
    attn_kernel<<<dim3(TT / 64, NH, 32), 128, 0, stream>>>(qkvb, ybuf);
    gemm_kernel<1, 0><<<dim3(NROWS / 128, 2), 256, 0, stream>>>(
        ybuf, woT + (size_t)l * 65536, bo + (size_t)l * CDIM,
        x, x, NROWS, CDIM, CDIM);
    ln_kernel<<<NROWS, 256, 0, stream>>>(x, ln2g + l * CDIM, ln2b + l * CDIM, hbuf);
    gemm_kernel<2, 1><<<dim3(NROWS / 128, 8), 256, 0, stream>>>(
        hbuf, wfcT + (size_t)l * 262144, bfc + (size_t)l * 1024,
        nullptr, ffb, NROWS, CDIM, 1024);
    gemm_kernel<1, 0><<<dim3(NROWS / 128, 2), 256, 0, stream>>>(
        ffb, wprT + (size_t)l * 262144, bpr + (size_t)l * CDIM,
        x, x, NROWS, 1024, CDIM);
  }

  ln_kernel<<<NROWS, 256, 0, stream>>>(x, lnfg, lnfb, hbuf);
  gemm_kernel<0, 0><<<dim3(NROWS / 128, 1), 256, 0, stream>>>(
      hbuf, wlmT, nullptr, nullptr, out, NROWS, CDIM, 128);
}
